// Net_16690242912867
// MI455X (gfx1250) — matrix operand order repaired and checked by reading, not compiled, not run on hardware
//
#include <hip/hip_runtime.h>
#include <hip/hip_bf16.h>

typedef __attribute__((ext_vector_type(16))) _Float16 v16h;
typedef __attribute__((ext_vector_type(8)))  float    v8f;

#define HID 128
#define FIN 32
#define NLAYER 3
#define BN_EPS 1e-5f

// ---------------------------------------------------------------------------
// CDNA5 async global->LDS copy (ASYNCcnt-tracked), CDNA5 ISA
// ---------------------------------------------------------------------------
__device__ __forceinline__ void async_load_b128(unsigned lds_off,
                                                const void* gaddr) {
    // GV mode: 64-bit vaddr, LDS byte address in VDST vgpr.
    asm volatile("global_load_async_to_lds_b128 %0, %1, off"
                 :
                 : "v"(lds_off), "v"((unsigned long long)(uintptr_t)gaddr)
                 : "memory");
}
__device__ __forceinline__ void wait_async_le8() {
    asm volatile("s_wait_asynccnt 0x8" ::: "memory");
}
__device__ __forceinline__ void wait_async_0() {
    asm volatile("s_wait_asynccnt 0x0" ::: "memory");
}
__device__ __forceinline__ unsigned lds_offset(const void* p) {
    // generic LDS pointer: addr[31:0] is the LDS byte address
    return (unsigned)(uintptr_t)p;
}

// ---------------------------------------------------------------------------
// WMMA fragment builders (CDNA5 wave32 layouts, CDNA5 ISA §7.12.2)
// ---------------------------------------------------------------------------
__device__ __forceinline__ v16h load_A_frag(const _Float16* tile, int ld, int lane) {
    int g = (lane >> 4) & 1;
    int m = lane & 15;
    v16h a;
#pragma unroll
    for (int v = 0; v < 8; ++v) {
        int kbase = ((v < 4) ? (2 * v) : (16 + 2 * (v - 4))) + 8 * g;
        a[2 * v + 0] = tile[m * ld + kbase + 0];
        a[2 * v + 1] = tile[m * ld + kbase + 1];
    }
    return a;
}

__device__ __forceinline__ v16h load_B_frag(const _Float16* tile, int ld, int lane) {
    int g = (lane >> 4) & 1;
    int n = lane & 15;
    v16h b;
#pragma unroll
    for (int i = 0; i < 16; ++i) b[i] = tile[(8 * g + i + ((i < 8) ? 0 : 8)) * ld + n];
    return b;
}

__device__ __forceinline__ v8f wmma_f16(v16h a, v16h b, v8f c) {
    return __builtin_amdgcn_wmma_f32_16x16x32_f16(false, a, false, b, (short)0, c,
                                                  false, false);
}

// ---------------------------------------------------------------------------
// Utility kernels
// ---------------------------------------------------------------------------
__global__ void zero_kernel(float* __restrict__ p, int n) {
    int i = blockIdx.x * blockDim.x + threadIdx.x;
    if (i < n) p[i] = 0.0f;
}

// Pre-convert all layer weights to f16 once: wf16[l][0]=W1_l, wf16[l][1]=W2_l
__global__ void w2h_kernel(const float* __restrict__ W1,
                           const float* __restrict__ W2,
                           _Float16* __restrict__ wf16) {
    const int per = HID * HID;
    int idx = blockIdx.x * blockDim.x + threadIdx.x;
    if (idx >= NLAYER * per) return;
    int l = idx / per, i = idx - l * per;
    wf16[(2 * l + 0) * per + i] = (_Float16)W1[idx];
    wf16[(2 * l + 1) * per + i] = (_Float16)W2[idx];
}

// ---------------------------------------------------------------------------
// Node encoder: h = x @ Wx + bx     ([N,32] @ [32,128]); K=32 = one WMMA
// ---------------------------------------------------------------------------
__global__ __launch_bounds__(256) void encode_kernel(
    const float* __restrict__ x, const float* __restrict__ Wx,
    const float* __restrict__ bx, float* __restrict__ h, int N) {
    __shared__ _Float16 sX[16 * FIN];
    __shared__ _Float16 sW[FIN * HID];
    int tid = threadIdx.x;
    int r0 = blockIdx.x * 16;
    for (int i = tid; i < 16 * FIN; i += 256) {
        int m = i / FIN, k = i % FIN;
        sX[i] = (_Float16)x[(size_t)(r0 + m) * FIN + k];
    }
    for (int i = tid; i < FIN * HID; i += 256) sW[i] = (_Float16)Wx[i];
    __syncthreads();

    int wave = tid >> 5, lane = tid & 31;
    int n0 = wave * 16;
    v16h a = load_A_frag(sX, FIN, lane);
    v16h b = load_B_frag(sW + n0, HID, lane);
    v8f acc = {};
    acc = wmma_f16(a, b, acc);

    int n = lane & 15, g = lane >> 4;
    float bias = bx[n0 + n];
#pragma unroll
    for (int r = 0; r < 8; ++r) {
        int m = r + 8 * g;
        h[(size_t)(r0 + m) * HID + n0 + n] = acc[r] + bias;
    }
}

// ---------------------------------------------------------------------------
// Message pass: aggr[dst] += relu(h[src] + edge_table[edge_attr])
// One wave per edge; indices forced scalar (SMEM), each lane moves 16B.
// ---------------------------------------------------------------------------
__global__ __launch_bounds__(256) void msg_kernel(
    const float* __restrict__ h, const float* __restrict__ etab,
    const int* __restrict__ eidx, const int* __restrict__ eattr,
    float* __restrict__ aggr, int E) {
    int gtid = blockIdx.x * 256 + threadIdx.x;
    int e = __builtin_amdgcn_readfirstlane(gtid >> 5);  // wave-uniform
    if (e >= E) return;
    int lane = threadIdx.x & 31;
    int src = eidx[e];
    int dst = eidx[E + e];
    int t = eattr[e];

    const float4 hv = *(const float4*)(h + (size_t)src * HID + lane * 4);
    const float4 ev = *(const float4*)(etab + (size_t)t * HID + lane * 4);
    float m0 = fmaxf(hv.x + ev.x, 0.0f);
    float m1 = fmaxf(hv.y + ev.y, 0.0f);
    float m2 = fmaxf(hv.z + ev.z, 0.0f);
    float m3 = fmaxf(hv.w + ev.w, 0.0f);
    float* ap = aggr + (size_t)dst * HID + lane * 4;
    atomicAdd(ap + 0, m0);
    atomicAdd(ap + 1, m1);
    atomicAdd(ap + 2, m2);
    atomicAdd(ap + 3, m3);
}

// ---------------------------------------------------------------------------
// Fused GINE MLP: z = relu(relu((h+aggr)@W1+b1)@W2+b2) + BN partial stats.
// W1/W2 streamed into LDS via async B128 copies: W1 waited before GEMM1,
// W2 arrives underneath GEMM1 (in-order ASYNCcnt completion).
// ---------------------------------------------------------------------------
__global__ __launch_bounds__(256) void mlp_kernel(
    const float* __restrict__ h, const float* __restrict__ aggr,
    const _Float16* __restrict__ w1h, const float* __restrict__ b1,
    const _Float16* __restrict__ w2h, const float* __restrict__ b2,
    float* __restrict__ z, float* __restrict__ bnsum, float* __restrict__ bnsq,
    int N) {
    const int LDT = HID + 8;  // padded LDS stride (halves)
    __shared__ _Float16 sT[32 * (HID + 8)];
    __shared__ _Float16 sW1[HID * (HID + 8)];
    __shared__ _Float16 sW2[HID * (HID + 8)];
    __shared__ _Float16 sU[32 * (HID + 8)];

    int tid = threadIdx.x;
    int r0 = blockIdx.x * 32;

    // ---- issue async weight copies: 16B chunks, 2048 per matrix ----
    // row k, chunk cc -> 8 halves at global k*128+cc*8, LDS k*LDT+cc*8
#pragma unroll
    for (int it = 0; it < 8; ++it) {
        int c = tid + it * 256;
        int k = c >> 4, cc = c & 15;
        async_load_b128(lds_offset(&sW1[k * LDT + cc * 8]), w1h + k * HID + cc * 8);
    }
#pragma unroll
    for (int it = 0; it < 8; ++it) {
        int c = tid + it * 256;
        int k = c >> 4, cc = c & 15;
        async_load_b128(lds_offset(&sW2[k * LDT + cc * 8]), w2h + k * HID + cc * 8);
    }

    // ---- load t = h + aggr tile (32x128) while weights fly ----
    for (int i4 = tid; i4 < 32 * HID / 4; i4 += 256) {
        int m = i4 >> 5, c4 = i4 & 31;
        size_t gi = (size_t)(r0 + m) * HID + c4 * 4;
        const float4 hv = *(const float4*)(h + gi);
        const float4 av = *(const float4*)(aggr + gi);
        _Float16* d = &sT[m * LDT + c4 * 4];
        d[0] = (_Float16)(hv.x + av.x);
        d[1] = (_Float16)(hv.y + av.y);
        d[2] = (_Float16)(hv.z + av.z);
        d[3] = (_Float16)(hv.w + av.w);
    }

    wait_async_le8();   // W1's 8 per-wave async loads complete (in-order)
    __syncthreads();    // all waves' W1 chunks + sT visible

    int wave = tid >> 5, lane = tid & 31;
    int n0 = wave * 16;
    int n = lane & 15, g = lane >> 4;

    // ---- GEMM1: u = relu(t @ W1 + b1) ----
    v8f acc0 = {}, acc1 = {};
#pragma unroll
    for (int k0 = 0; k0 < HID; k0 += 32) {
        v16h b = load_B_frag(sW1 + k0 * LDT + n0, LDT, lane);
        v16h a0 = load_A_frag(sT + k0, LDT, lane);
        v16h a1 = load_A_frag(sT + 16 * LDT + k0, LDT, lane);
        acc0 = wmma_f16(a0, b, acc0);
        acc1 = wmma_f16(a1, b, acc1);
    }
    float bias1 = b1[n0 + n];
#pragma unroll
    for (int r = 0; r < 8; ++r) {
        float u0 = fmaxf(acc0[r] + bias1, 0.0f);
        float u1 = fmaxf(acc1[r] + bias1, 0.0f);
        sU[(r + 8 * g) * LDT + n0 + n] = (_Float16)u0;
        sU[(16 + r + 8 * g) * LDT + n0 + n] = (_Float16)u1;
    }

    wait_async_0();     // W2 resident
    __syncthreads();    // sU + all W2 chunks visible

    // ---- GEMM2: z = relu(u @ W2 + b2) ----
    acc0 = (v8f){};
    acc1 = (v8f){};
#pragma unroll
    for (int k0 = 0; k0 < HID; k0 += 32) {
        v16h b = load_B_frag(sW2 + k0 * LDT + n0, LDT, lane);
        v16h a0 = load_A_frag(sU + k0, LDT, lane);
        v16h a1 = load_A_frag(sU + 16 * LDT + k0, LDT, lane);
        acc0 = wmma_f16(a0, b, acc0);
        acc1 = wmma_f16(a1, b, acc1);
    }
    float bias2 = b2[n0 + n];
    float s = 0.0f, sq = 0.0f;
#pragma unroll
    for (int r = 0; r < 8; ++r) {
        float z0 = fmaxf(acc0[r] + bias2, 0.0f);
        float z1 = fmaxf(acc1[r] + bias2, 0.0f);
        z[(size_t)(r0 + r + 8 * g) * HID + n0 + n] = z0;
        z[(size_t)(r0 + 16 + r + 8 * g) * HID + n0 + n] = z1;
        s += z0 + z1;
        sq += z0 * z0 + z1 * z1;
    }
    s += __shfl_xor(s, 16, 32);
    sq += __shfl_xor(sq, 16, 32);
    if (g == 0) {
        atomicAdd(&bnsum[n0 + n], s);
        atomicAdd(&bnsq[n0 + n], sq);
    }
}

// ---------------------------------------------------------------------------
// BatchNorm normalize (batch stats): h = gamma*(z-mu)*rsqrt(var+eps)+beta
// ---------------------------------------------------------------------------
__global__ void bn_kernel(const float* __restrict__ z,
                          const float* __restrict__ bnsum,
                          const float* __restrict__ bnsq,
                          const float* __restrict__ gamma,
                          const float* __restrict__ beta,
                          float* __restrict__ h, int N) {
    int i4 = blockIdx.x * blockDim.x + threadIdx.x;
    if (i4 >= N * HID / 4) return;
    int c0 = (i4 * 4) & 127;
    float invN = 1.0f / (float)N;
    const float4 v = *(const float4*)(z + (size_t)i4 * 4);
    float4 o;
    float* ov = &o.x;
    const float* vv = &v.x;
#pragma unroll
    for (int j = 0; j < 4; ++j) {
        int c = c0 + j;
        float mu = bnsum[c] * invN;
        float var = bnsq[c] * invN - mu * mu;
        ov[j] = gamma[c] * (vv[j] - mu) * rsqrtf(var + BN_EPS) + beta[c];
    }
    *(float4*)(h + (size_t)i4 * 4) = o;
}

// ---------------------------------------------------------------------------
// Pooling: segment sum/count per graph, then divide
// ---------------------------------------------------------------------------
__global__ __launch_bounds__(256) void pool_kernel(
    const float* __restrict__ h, const int* __restrict__ batch,
    float* __restrict__ pooled, float* __restrict__ counts, int N) {
    int gtid = blockIdx.x * 256 + threadIdx.x;
    int node = __builtin_amdgcn_readfirstlane(gtid >> 5);  // wave-uniform
    if (node >= N) return;
    int lane = threadIdx.x & 31;
    int b = batch[node];
    const float4 hv = *(const float4*)(h + (size_t)node * HID + lane * 4);
    float* pp = pooled + (size_t)b * HID + lane * 4;
    atomicAdd(pp + 0, hv.x);
    atomicAdd(pp + 1, hv.y);
    atomicAdd(pp + 2, hv.z);
    atomicAdd(pp + 3, hv.w);
    if (lane == 0) atomicAdd(&counts[b], 1.0f);
}

__global__ void finalize_kernel(const float* __restrict__ pooled,
                                const float* __restrict__ counts,
                                float* __restrict__ out, int G) {
    int idx = blockIdx.x * blockDim.x + threadIdx.x;
    if (idx >= G * HID) return;
    int g = idx >> 7;
    out[idx] = pooled[idx] / fmaxf(counts[g], 1.0f);
}

// ---------------------------------------------------------------------------
// Launcher
// ---------------------------------------------------------------------------
extern "C" void kernel_launch(void* const* d_in, const int* in_sizes, int n_in,
                              void* d_out, int out_size, void* d_ws, size_t ws_size,
                              hipStream_t stream) {
    const float* x     = (const float*)d_in[0];
    const float* Wx    = (const float*)d_in[1];
    const float* bx    = (const float*)d_in[2];
    const float* etab  = (const float*)d_in[3];
    const float* W1    = (const float*)d_in[4];
    const float* b1    = (const float*)d_in[5];
    const float* W2    = (const float*)d_in[6];
    const float* b2    = (const float*)d_in[7];
    const float* gamma = (const float*)d_in[8];
    const float* beta  = (const float*)d_in[9];
    const int* eidx    = (const int*)d_in[10];
    const int* eattr   = (const int*)d_in[11];
    const int* batch   = (const int*)d_in[12];

    const int E = in_sizes[11];
    const int N = in_sizes[12];
    const int G = out_size / HID;
    const size_t NH = (size_t)N * HID;
    const int per = HID * HID;

    float* ws     = (float*)d_ws;
    float* h      = ws;                // N*H
    float* aggr   = h + NH;            // N*H
    float* z      = aggr + NH;         // N*H
    float* bnsum  = z + NH;            // H
    float* bnsq   = bnsum + HID;       // H
    float* pooled = bnsq + HID;        // G*H
    float* counts = pooled + (size_t)G * HID;          // G
    _Float16* wf16 = (_Float16*)(counts + G);          // NLAYER*2*H*H (16B aligned)

    // zero pool accumulators; convert weights to f16 (once per launch)
    {
        int n = G * HID + G;
        zero_kernel<<<(n + 255) / 256, 256, 0, stream>>>(pooled, n);
        w2h_kernel<<<(NLAYER * per + 255) / 256, 256, 0, stream>>>(W1, W2, wf16);
    }

    // node encoder
    encode_kernel<<<(N + 15) / 16, 256, 0, stream>>>(x, Wx, bx, h, N);

    for (int l = 0; l < NLAYER; ++l) {
        zero_kernel<<<((int)NH + 255) / 256, 256, 0, stream>>>(aggr, (int)NH);
        msg_kernel<<<(E * 32 + 255) / 256, 256, 0, stream>>>(h, etab, eidx, eattr,
                                                             aggr, E);
        zero_kernel<<<1, 256, 0, stream>>>(bnsum, 2 * HID);
        mlp_kernel<<<(N + 31) / 32, 256, 0, stream>>>(
            h, aggr, wf16 + (size_t)(2 * l) * per, b1 + (size_t)l * HID,
            wf16 + (size_t)(2 * l + 1) * per, b2 + (size_t)l * HID, z, bnsum, bnsq,
            N);
        bn_kernel<<<((int)(NH / 4) + 255) / 256, 256, 0, stream>>>(
            z, bnsum, bnsq, gamma + (size_t)l * HID, beta + (size_t)l * HID, h, N);
    }

    // global mean pool
    pool_kernel<<<(N * 32 + 255) / 256, 256, 0, stream>>>(h, batch, pooled, counts, N);
    finalize_kernel<<<(G * HID + 255) / 256, 256, 0, stream>>>(pooled, counts,
                                                               (float*)d_out, G);
}
